// StructureAwareDynamicVQ_67619965108645
// MI455X (gfx1250) — matrix operand order repaired and checked by reading, not compiled, not run on hardware
//
#include <hip/hip_runtime.h>
#include <hip/hip_bf16.h>

// MI455X / gfx1250 implementation of StructureAwareDynamicVQ.
// Distance GEMM (131072x64 @ 64x1024 and @64x16) via v_wmma_f32_16x16x32_f16.
// Memory-bound problem (~200MB traffic @ 23.3TB/s); single pass over inputs
// for the quantize phase, one extra read for the loss/write phase.

typedef _Float16 f16;
typedef __attribute__((ext_vector_type(16))) _Float16 v16h;
typedef __attribute__((ext_vector_type(8)))  _Float16 v8h;
typedef __attribute__((ext_vector_type(8)))  float    v8f;

#define NSHAPE 1024
#define NCOLOR 16
#define HALF   64
#define CCH    128
#define HWW    4096
#define TILE   128
#define NUMEL  16777216   // 32*128*64*64
#define NPOS   131072

// workspace layout (bytes); requires ws_size >= 137472
#define WS_ACC   0        // float[4]: [0]=sse, [1]=rep_shape, [2]=rep_color
#define WS_SSQ   64       // float[1024]  ||w||^2 shape
#define WS_CSQ   4224     // float[16]    ||w||^2 color
#define WS_WS16  4352     // f16[65536]   W_shape as f16
#define WS_WC16  135424   // f16[1024]    W_color as f16

// d_out layout (floats)
#define O_VQ   16777216
#define O_REP  16777217
#define O_SIDX 16777218
#define O_CIDX (16777218 + 131072)

// dynamic LDS partition (bytes)
#define L_X    0          // f16[128*136]  staged X tile (pos-major, stride 136)
#define L_Q    34816      // f32[128*132]  q tile (ch-major, stride 132)
#define L_RED  102400     // f32[256]
#define L_TOT  103424

union Fr { v16h v; v8h h[2]; };

__global__ void vq_zero(char* ws) {
    if (threadIdx.x < 4) ((float*)(ws + WS_ACC))[threadIdx.x] = 0.0f;
}

__global__ void vq_prep(const float* __restrict__ Ws, const float* __restrict__ Wc,
                        char* __restrict__ ws) {
    const int tid = blockIdx.x * 256 + threadIdx.x;
    f16*   Ws16 = (f16*)(ws + WS_WS16);
    f16*   Wc16 = (f16*)(ws + WS_WC16);
    float* ssq  = (float*)(ws + WS_SSQ);
    float* csq  = (float*)(ws + WS_CSQ);
    if (tid < NSHAPE) {
        float s = 0.0f;
        #pragma unroll 8
        for (int d = 0; d < HALF; ++d) {
            float v = Ws[tid * HALF + d];
            Ws16[tid * HALF + d] = (f16)v;
            s += v * v;
        }
        ssq[tid] = s;
    } else if (tid < NSHAPE + NCOLOR) {
        const int c = tid - NSHAPE;
        float s = 0.0f;
        #pragma unroll 8
        for (int d = 0; d < HALF; ++d) {
            float v = Wc[c * HALF + d];
            Wc16[c * HALF + d] = (f16)v;
            s += v * v;
        }
        csq[c] = s;
    }
}

__global__ __launch_bounds__(256) void vq_main(const float* __restrict__ inp,
                                               const float* __restrict__ Wsf,
                                               const float* __restrict__ Wcf,
                                               char* __restrict__ ws,
                                               float* __restrict__ out) {
    extern __shared__ char smem[];
    f16*   lx  = (f16*)(smem + L_X);
    float* qd  = (float*)(smem + L_Q);
    float* red = (float*)(smem + L_RED);

    const int  t    = threadIdx.x;
    const long base = (long)blockIdx.x * TILE;          // global position base
    const long bimg = base >> 12;                        // image index (b)
    const long hw0  = base & 4095;
    const long ibase = bimg * (long)CCH * HWW + hw0;     // &inputs[b][0][hw0]

    // ---- phase 0: stage X tile into LDS as f16 (pos-major, stride 136) ----
    {
        const int pos = t & 127;
        const int ch0 = t >> 7;
        #pragma unroll 4
        for (int c2 = 0; c2 < 64; ++c2) {
            const int ch = c2 * 2 + ch0;
            float v = inp[ibase + (long)ch * HWW + pos];
            lx[pos * 136 + ch] = (f16)v;
        }
    }
    __syncthreads();

    const int wave = t >> 5;
    const int lane = t & 31;
    const int lr   = lane & 15;   // row M within tile (A) / column N (B,C,D)
    const int g    = lane >> 4;   // K-half selector

    // ---- A fragments: per lane two contiguous 8xf16 chunks per K-step ----
    Fr as0, as1, ac0, ac1;
    {
        const f16* rp = &lx[(wave * 16 + lr) * 136 + g * 16];
        as0.h[0] = *(const v8h*)(rp +   0); as0.h[1] = *(const v8h*)(rp +   8);
        as1.h[0] = *(const v8h*)(rp +  32); as1.h[1] = *(const v8h*)(rp +  40);
        ac0.h[0] = *(const v8h*)(rp +  64); ac0.h[1] = *(const v8h*)(rp +  72);
        ac1.h[0] = *(const v8h*)(rp +  96); ac1.h[1] = *(const v8h*)(rp + 104);
    }

    const f16*   Ws16 = (const f16*)(ws + WS_WS16);
    const f16*   Wc16 = (const f16*)(ws + WS_WC16);
    const float* ssq  = (const float*)(ws + WS_SSQ);
    const float* csq  = (const float*)(ws + WS_CSQ);

    float bv[8], bvc[8];
    int   bi[8], bic[8];
    #pragma unroll
    for (int r = 0; r < 8; ++r) { bv[r] = 3.4e38f; bi[r] = 0; bvc[r] = 3.4e38f; bic[r] = 0; }

    // ---- shape codebook: 64 N-tiles x 2 WMMA (K=32) ----
    #pragma unroll 2
    for (int nt = 0; nt < 64; ++nt) {
        const int n = nt * 16 + lr;
        const f16* bp = &Ws16[n * HALF + g * 16];
        Fr bs0, bs1;
        bs0.h[0] = *(const v8h*)(bp +  0); bs0.h[1] = *(const v8h*)(bp +  8);
        bs1.h[0] = *(const v8h*)(bp + 32); bs1.h[1] = *(const v8h*)(bp + 40);
        v8f acc = {0.f, 0.f, 0.f, 0.f, 0.f, 0.f, 0.f, 0.f};
        acc = __builtin_amdgcn_wmma_f32_16x16x32_f16(false, as0.v, false, bs0.v,
                                                     (short)0, acc, false, false);
        acc = __builtin_amdgcn_wmma_f32_16x16x32_f16(false, as1.v, false, bs1.v,
                                                     (short)0, acc, false, false);
        const float wsq = ssq[n];
        #pragma unroll
        for (int r = 0; r < 8; ++r) {
            float cand = wsq - 2.0f * acc[r];
            if (cand < bv[r]) { bv[r] = cand; bi[r] = n; }
        }
    }

    // ---- color codebook: single N-tile ----
    {
        const int n = lr;
        const f16* bp = &Wc16[n * HALF + g * 16];
        Fr bc0, bc1;
        bc0.h[0] = *(const v8h*)(bp +  0); bc0.h[1] = *(const v8h*)(bp +  8);
        bc1.h[0] = *(const v8h*)(bp + 32); bc1.h[1] = *(const v8h*)(bp + 40);
        v8f acc = {0.f, 0.f, 0.f, 0.f, 0.f, 0.f, 0.f, 0.f};
        acc = __builtin_amdgcn_wmma_f32_16x16x32_f16(false, ac0.v, false, bc0.v,
                                                     (short)0, acc, false, false);
        acc = __builtin_amdgcn_wmma_f32_16x16x32_f16(false, ac1.v, false, bc1.v,
                                                     (short)0, acc, false, false);
        const float wsq = csq[n];
        #pragma unroll
        for (int r = 0; r < 8; ++r) {
            float cand = wsq - 2.0f * acc[r];
            if (cand < bvc[r]) { bvc[r] = cand; bic[r] = n; }
        }
    }

    // ---- cross-lane argmin within each 16-lane half (C/D column group) ----
    #pragma unroll
    for (int m = 1; m <= 8; m <<= 1) {
        #pragma unroll
        for (int r = 0; r < 8; ++r) {
            float ov  = __shfl_xor(bv[r],  m, 32);
            int   oi  = __shfl_xor(bi[r],  m, 32);
            if (ov < bv[r] || (ov == bv[r] && oi < bi[r])) { bv[r] = ov; bi[r] = oi; }
            float ovc = __shfl_xor(bvc[r], m, 32);
            int   oic = __shfl_xor(bic[r], m, 32);
            if (ovc < bvc[r] || (ovc == bvc[r] && oic < bic[r])) { bvc[r] = ovc; bic[r] = oic; }
        }
    }

    // ---- write indices (as float, per d_out dtype) ----
    if (lr < 8) {
        const long pos = base + wave * 16 + lr + 8 * g;   // C/D row M = r + 8*g
        out[O_SIDX + pos] = (float)bi[lr];
        out[O_CIDX + pos] = (float)bic[lr];
    }

    // ---- gather f32 codebook rows into LDS q tile (ch-major, stride 132) ----
    #pragma unroll
    for (int r = 0; r < 8; ++r) {
        const int pl = wave * 16 + r + 8 * g;   // local position of row M
        const int so = bi[r]  * HALF;
        const int co = bic[r] * HALF;
        #pragma unroll
        for (int k = 0; k < 4; ++k) {
            const int ch = lr + 16 * k;
            qd[ch * 132 + pl]        = Wsf[so + ch];
            qd[(HALF + ch) * 132 + pl] = Wcf[co + ch];
        }
    }
    __syncthreads();

    // ---- phase 3: coalesced write of quantized + SSE accumulation ----
    float sse = 0.0f;
    {
        const int pos = t & 127;
        const int ch0 = t >> 7;
        #pragma unroll 4
        for (int c2 = 0; c2 < 64; ++c2) {
            const int ch = c2 * 2 + ch0;
            const long ga = ibase + (long)ch * HWW + pos;
            float q = qd[ch * 132 + pos];
            float x = inp[ga];
            out[ga] = q;                 // quantized == q value-wise
            float d = q - x;
            sse += d * d;
        }
    }
    red[t] = sse;
    __syncthreads();
    for (int s = 128; s > 0; s >>= 1) {
        if (t < s) red[t] += red[t + s];
        __syncthreads();
    }
    if (t == 0) atomicAdd((float*)(ws + WS_ACC), red[0]);
}

__global__ __launch_bounds__(256) void vq_rep(const float* __restrict__ Ws,
                                              const float* __restrict__ Wc,
                                              char* __restrict__ ws) {
    __shared__ float red[256];
    const int t   = threadIdx.x;
    const int blk = blockIdx.x;
    const float* ssq = (const float*)(ws + WS_SSQ);
    const float* csq = (const float*)(ws + WS_CSQ);
    float s = 0.0f;
    if (blk < NSHAPE) {
        const int i = blk;
        const float sqi = ssq[i];
        const float* wi = &Ws[i * HALF];
        for (int j = t; j < NSHAPE; j += 256) {
            const float* wj = &Ws[j * HALF];
            float dot = 0.0f;
            #pragma unroll 16
            for (int d = 0; d < HALF; ++d) dot += wi[d] * wj[d];
            float dd = sqi + ssq[j] - 2.0f * dot;
            if (i == j) dd += 1e-5f;
            s += 1.0f / (dd + 1e-4f);
        }
    } else {
        const int i = blk - NSHAPE;
        if (t < NCOLOR) {
            const float sqi = csq[i];
            const float* wi = &Wc[i * HALF];
            const float* wj = &Wc[t * HALF];
            float dot = 0.0f;
            #pragma unroll 16
            for (int d = 0; d < HALF; ++d) dot += wi[d] * wj[d];
            float dd = sqi + csq[t] - 2.0f * dot;
            if (i == t) dd += 1e-5f;
            s += 1.0f / (dd + 1e-4f);
        }
    }
    red[t] = s;
    __syncthreads();
    for (int k = 128; k > 0; k >>= 1) {
        if (t < k) red[t] += red[t + k];
        __syncthreads();
    }
    if (t == 0) {
        float* acc = (float*)(ws + WS_ACC);
        atomicAdd(blk < NSHAPE ? (acc + 1) : (acc + 2), red[0]);
    }
}

__global__ void vq_fin(const char* __restrict__ ws, float* __restrict__ out) {
    const float* acc = (const float*)(ws + WS_ACC);
    float vq   = acc[0] * (1.0f + 0.25f) / (float)NUMEL;  // q_latent + 0.25*e_latent
    float reps = (acc[1] - (float)NSHAPE * 1e4f) / ((float)NSHAPE * (float)(NSHAPE - 1));
    float repc = (acc[2] - (float)NCOLOR * 1e4f) / ((float)NCOLOR * (float)(NCOLOR - 1));
    out[O_VQ]  = vq;
    out[O_REP] = (reps + repc) * 0.1f;
}

extern "C" void kernel_launch(void* const* d_in, const int* in_sizes, int n_in,
                              void* d_out, int out_size, void* d_ws, size_t ws_size,
                              hipStream_t stream) {
    const float* inp = (const float*)d_in[0];   // (32,128,64,64) f32
    const float* Ws  = (const float*)d_in[1];   // (1024,64) f32
    const float* Wc  = (const float*)d_in[2];   // (16,64) f32
    float* out = (float*)d_out;
    char*  ws  = (char*)d_ws;
    (void)in_sizes; (void)n_in; (void)out_size; (void)ws_size;

    vq_zero<<<1, 32, 0, stream>>>(ws);
    vq_prep<<<5, 256, 0, stream>>>(Ws, Wc, ws);
    vq_main<<<NPOS / TILE, 256, L_TOT, stream>>>(inp, Ws, Wc, ws, out);
    vq_rep<<<NSHAPE + NCOLOR, 256, 0, stream>>>(Ws, Wc, ws);
    vq_fin<<<1, 1, 0, stream>>>(ws, out);
}
